// QNetwork_57406532878787
// MI455X (gfx1250) — matrix operand order repaired and checked by reading, not compiled, not run on hardware
//
#include <hip/hip_runtime.h>

// ---------------------------------------------------------------------------
// CDNA5 (gfx1250, wave32) fused GRU + MLP Q-network.
// GEMMs: v_wmma_f32_16x16x32_f16, LDS double-buffered via
// global_load_async_to_lds_b128 (ASYNCcnt). XOR-swizzled LDS, K-tile 64.
// ---------------------------------------------------------------------------

typedef __attribute__((ext_vector_type(16))) _Float16 v16h;
typedef __attribute__((ext_vector_type(8)))  _Float16 v8h;   // 16-byte chunk
typedef __attribute__((ext_vector_type(8)))  float    v8f;

#define BATCH  65536
#define IN_DIM 320
#define GRU    512
#define HID    1024
#define KT     64     // K-halves staged per LDS buffer (2 wmma sub-steps)

// Workspace layout (in _Float16 elements). Total = 158,040,064 halves = 316 MB.
#define OFF_WIH 0ull            // 1536*320
#define OFF_WHH 491520ull       // 1536*512
#define OFF_W1  1277952ull      // 1024*512
#define OFF_W2  1802240ull      // 1024*1024
#define OFF_SA  2850816ull      // 65536*320  (concat(state,action) f16)
#define OFF_H   23822336ull     // 65536*512  (hidden f16)
#define OFF_HN  57376768ull     // 65536*512  (h_new f16)
#define OFF_X1  90931200ull     // 65536*1024 (relu(fc1) f16)

// ---- CDNA5 async copy: global -> LDS, 16 bytes per lane, ASYNCcnt-tracked --
__device__ __forceinline__ void async_cp16(unsigned lds_byte,
                                           unsigned long long gaddr) {
  asm volatile("global_load_async_to_lds_b128 %0, %1, off"
               :: "v"(lds_byte), "v"(gaddr) : "memory");
}
__device__ __forceinline__ void async_wait0() {
  asm volatile("s_wait_asynccnt 0x0" ::: "memory");
}

// XOR-swizzled LDS offset (halves) for 16-byte chunk `seg` (0..7) of `row`.
// Bank start (row*32 + s'*4) mod 64 is distinct across any 16-row fragment
// read -> conflict-free without padding.
__device__ __forceinline__ int swz_off(int row, int seg) {
  return row * KT + ((seg ^ ((row >> 1) & 7)) << 3);
}

// ---- fragment loaders from swizzled LDS (CDNA5 wave32 layouts, ISA 7.12.2) -
// A 16x32 f16: lanes 0-15 rows 0-15 K{0..7,16..23}; lanes 16-31 K{8..15,24..31}
__device__ __forceinline__ v16h lds_frag_a(const _Float16* lbuf, int rt, int ks,
                                           int lane) {
  const int row = rt * 16 + (lane & 15);
  const int s0  = ks * 4 + (lane >> 4);
  v16h a;
  ((v8h*)&a)[0] = *(const v8h*)(lbuf + swz_off(row, s0));
  ((v8h*)&a)[1] = *(const v8h*)(lbuf + swz_off(row, s0 + 2));
  return a;
}
// B 32x16 f16: lane n = column n, K{0..7,16..23}; lane n+16 = column n, K{8..15,24..31}.
__device__ __forceinline__ v16h lds_frag_b(const _Float16* lbuf, int rt, int ks,
                                           int lane) {
  const int row = rt * 16 + (lane & 15);
  const int s0  = ks * 4 + (lane >> 4);
  v16h b;
  ((v8h*)&b)[0] = *(const v8h*)(lbuf + swz_off(row, s0));
  ((v8h*)&b)[1] = *(const v8h*)(lbuf + swz_off(row, s0 + 2));
  return b;
}

__device__ __forceinline__ v8f wmma(v16h a, v16h b, v8f c) {
  return __builtin_amdgcn_wmma_f32_16x16x32_f16(false, a, false, b,
                                                (short)0, c, false, false);
}
__device__ __forceinline__ float sigmoidf_(float x) {
  return 1.0f / (1.0f + __expf(-x));
}

// ---- small prep kernels ----------------------------------------------------
__global__ void cvt_f32_f16(const float* __restrict__ s,
                            _Float16* __restrict__ d, size_t n) {
  size_t i = (size_t)blockIdx.x * blockDim.x + threadIdx.x;
  if (i < n) d[i] = (_Float16)s[i];
}

__global__ void build_sa16(const float* __restrict__ st,
                           const float* __restrict__ ac,
                           _Float16* __restrict__ d) {
  size_t i = (size_t)blockIdx.x * blockDim.x + threadIdx.x;
  if (i >= (size_t)BATCH * IN_DIM) return;
  int r = (int)(i / IN_DIM), c = (int)(i % IN_DIM);
  float v = (c < 256) ? st[(size_t)r * 256 + c] : ac[(size_t)r * 64 + (c - 256)];
  d[i] = (_Float16)v;
}

__global__ void init_q(float* __restrict__ q, const float* __restrict__ b3) {
  int i = blockIdx.x * blockDim.x + threadIdx.x;
  if (i < BATCH) q[i] = b3[0];
}

// ---- fused GRU cell --------------------------------------------------------
// Tile: 128 batch rows x 32 gate columns; 8 waves x 16 rows. 4 accumulators
// (r, z, i_n, h_n) x 2 column tiles. Two K phases (input 320, hidden 512).
__global__ __launch_bounds__(256)
void gru_kernel(const _Float16* __restrict__ ws,
                const float* __restrict__ hidden,
                const float* __restrict__ b_ih, const float* __restrict__ b_hh,
                float* __restrict__ h_out, _Float16* __restrict__ hn16) {
  __shared__ _Float16 sA[2][128 * KT];   // 32 KB
  __shared__ _Float16 sB[2][96 * KT];    // 24 KB (3 gates x 32 cols)
  const int tid  = threadIdx.x;
  const int lane = tid & 31;
  const int wave = tid >> 5;
  const int mblk = blockIdx.x * 128;
  const int n0   = blockIdx.y * 32;

  // staging assignment: thread owns chunks (row = tid/8 + 32*i, seg = tid%8)
  const int arow = tid >> 3;
  const int aseg = tid & 7;
  unsigned ldsA0[4], ldsB0[3];
#pragma unroll
  for (int i = 0; i < 4; ++i)
    ldsA0[i] = (unsigned)(unsigned long long)(&sA[0][swz_off(arow + 32 * i, aseg)]);
#pragma unroll
  for (int i = 0; i < 3; ++i)
    ldsB0[i] = (unsigned)(unsigned long long)(&sB[0][swz_off(arow + 32 * i, aseg)]);

  v8f acc_r[2] = {}, acc_z[2] = {}, acc_in[2] = {}, acc_hn[2] = {};

  auto run_phase = [&](const _Float16* __restrict__ src,
                       const _Float16* __restrict__ wgt, int lda, int ntiles,
                       v8f* accN) {
    unsigned long long gA[4], gB[3];
#pragma unroll
    for (int i = 0; i < 4; ++i)
      gA[i] = (unsigned long long)(src + (size_t)(mblk + arow + 32 * i) * lda + aseg * 8);
#pragma unroll
    for (int i = 0; i < 3; ++i)   // B rows i*32+arow-32*i... gate i, col arow
      gB[i] = (unsigned long long)(wgt + (size_t)(i * GRU + n0 + arow) * lda + aseg * 8);
    auto stage = [&](int buf) {
      const unsigned oa = (unsigned)buf * (unsigned)sizeof(sA[0]);
      const unsigned ob = (unsigned)buf * (unsigned)sizeof(sB[0]);
#pragma unroll
      for (int i = 0; i < 4; ++i) { async_cp16(ldsA0[i] + oa, gA[i]); gA[i] += 2 * KT; }
#pragma unroll
      for (int i = 0; i < 3; ++i) { async_cp16(ldsB0[i] + ob, gB[i]); gB[i] += 2 * KT; }
    };
    stage(0);
    async_wait0();
    __syncthreads();
    for (int kt = 0; kt < ntiles; ++kt) {
      const int cur = kt & 1;
      if (kt + 1 < ntiles) stage(cur ^ 1);
#pragma unroll
      for (int ks = 0; ks < 2; ++ks) {
        v16h a  = lds_frag_a(sA[cur], wave, ks, lane);
        v16h b0 = lds_frag_b(sB[cur], 0, ks, lane);
        v16h b1 = lds_frag_b(sB[cur], 1, ks, lane);
        v16h b2 = lds_frag_b(sB[cur], 2, ks, lane);
        v16h b3 = lds_frag_b(sB[cur], 3, ks, lane);
        v16h b4 = lds_frag_b(sB[cur], 4, ks, lane);
        v16h b5 = lds_frag_b(sB[cur], 5, ks, lane);
        acc_r[0] = wmma(a, b0, acc_r[0]);
        acc_r[1] = wmma(a, b1, acc_r[1]);
        acc_z[0] = wmma(a, b2, acc_z[0]);
        acc_z[1] = wmma(a, b3, acc_z[1]);
        accN[0]  = wmma(a, b4, accN[0]);
        accN[1]  = wmma(a, b5, accN[1]);
      }
      async_wait0();
      __syncthreads();
    }
  };

  run_phase(ws + OFF_SA, ws + OFF_WIH, IN_DIM, IN_DIM / KT, acc_in); // gi
  run_phase(ws + OFF_H,  ws + OFF_WHH, GRU,    GRU / KT,    acc_hn); // gh

  // ---- epilogue: gates + h_new (f32 to d_out, f16 to ws for FC1) ----
  const int m0 = mblk + wave * 16;
  const int mo = (lane >> 4) * 8;
  const int nc = lane & 15;
#pragma unroll
  for (int t = 0; t < 2; ++t) {
    const int j = n0 + t * 16 + nc;
    const float brj = b_ih[j] + b_hh[j];
    const float bzj = b_ih[GRU + j] + b_hh[GRU + j];
    const float bij = b_ih[2 * GRU + j];
    const float bhj = b_hh[2 * GRU + j];
#pragma unroll
    for (int v = 0; v < 8; ++v) {
      const int m = m0 + mo + v;
      const float r = sigmoidf_(acc_r[t][v] + brj);
      const float z = sigmoidf_(acc_z[t][v] + bzj);
      const float n = tanhf(acc_in[t][v] + bij + r * (acc_hn[t][v] + bhj));
      const float ho = hidden[(size_t)m * GRU + j];
      const float hn = (1.0f - z) * n + z * ho;
      h_out[(size_t)m * GRU + j] = hn;
      hn16 [(size_t)m * GRU + j] = (_Float16)hn;
    }
  }
}

// ---- FC1: x1 = relu(h_new @ w1^T + b1), 65536x512x1024 ---------------------
// Workgroup 128x128, 8 waves in 4(M)x2(N): wave tile 32x64 = 2x4 wmma tiles.
__global__ __launch_bounds__(256)
void fc1_kernel(const _Float16* __restrict__ ws,
                const float* __restrict__ b1, _Float16* __restrict__ x1) {
  __shared__ _Float16 sA[2][128 * KT];   // 32 KB
  __shared__ _Float16 sB[2][128 * KT];   // 32 KB
  const int tid = threadIdx.x, lane = tid & 31, wave = tid >> 5;
  const int mblk = blockIdx.x * 128, nblk = blockIdx.y * 128;
  const _Float16* A = ws + OFF_HN;
  const _Float16* W = ws + OFF_W1;

  const int arow = tid >> 3;
  const int aseg = tid & 7;
  unsigned ldsA0[4], ldsB0[4];
#pragma unroll
  for (int i = 0; i < 4; ++i) {
    ldsA0[i] = (unsigned)(unsigned long long)(&sA[0][swz_off(arow + 32 * i, aseg)]);
    ldsB0[i] = (unsigned)(unsigned long long)(&sB[0][swz_off(arow + 32 * i, aseg)]);
  }
  unsigned long long gA[4], gB[4];
#pragma unroll
  for (int i = 0; i < 4; ++i) {
    gA[i] = (unsigned long long)(A + (size_t)(mblk + arow + 32 * i) * GRU + aseg * 8);
    gB[i] = (unsigned long long)(W + (size_t)(nblk + arow + 32 * i) * GRU + aseg * 8);
  }
  auto stage = [&](int buf) {
    const unsigned ob = (unsigned)buf * (unsigned)sizeof(sA[0]);
#pragma unroll
    for (int i = 0; i < 4; ++i) { async_cp16(ldsA0[i] + ob, gA[i]); gA[i] += 2 * KT; }
#pragma unroll
    for (int i = 0; i < 4; ++i) { async_cp16(ldsB0[i] + ob, gB[i]); gB[i] += 2 * KT; }
  };

  v8f acc[2][4] = {};
  stage(0);
  async_wait0();
  __syncthreads();
  const int art = (wave >> 1) * 2;
  const int bct = (wave & 1) * 4;
  for (int kt = 0; kt < GRU / KT; ++kt) {
    const int cur = kt & 1;
    if (kt + 1 < GRU / KT) stage(cur ^ 1);
#pragma unroll
    for (int ks = 0; ks < 2; ++ks) {
      v16h a0 = lds_frag_a(sA[cur], art + 0, ks, lane);
      v16h a1 = lds_frag_a(sA[cur], art + 1, ks, lane);
      v16h b0 = lds_frag_b(sB[cur], bct + 0, ks, lane);
      v16h b1 = lds_frag_b(sB[cur], bct + 1, ks, lane);
      v16h b2 = lds_frag_b(sB[cur], bct + 2, ks, lane);
      v16h b3 = lds_frag_b(sB[cur], bct + 3, ks, lane);
      acc[0][0] = wmma(a0, b0, acc[0][0]);
      acc[1][0] = wmma(a1, b0, acc[1][0]);
      acc[0][1] = wmma(a0, b1, acc[0][1]);
      acc[1][1] = wmma(a1, b1, acc[1][1]);
      acc[0][2] = wmma(a0, b2, acc[0][2]);
      acc[1][2] = wmma(a1, b2, acc[1][2]);
      acc[0][3] = wmma(a0, b3, acc[0][3]);
      acc[1][3] = wmma(a1, b3, acc[1][3]);
    }
    async_wait0();
    __syncthreads();
  }
  const int m0 = mblk + (wave >> 1) * 32;
  const int n0 = nblk + (wave & 1) * 64;
  const int mo = (lane >> 4) * 8;
  const int nc = lane & 15;
#pragma unroll
  for (int rt = 0; rt < 2; ++rt)
#pragma unroll
    for (int c = 0; c < 4; ++c) {
      const int n = n0 + c * 16 + nc;
      const float bb = b1[n];
#pragma unroll
      for (int v = 0; v < 8; ++v) {
        const int m = m0 + rt * 16 + mo + v;
        float y = acc[rt][c][v] + bb;
        x1[(size_t)m * HID + n] = (_Float16)(y > 0.0f ? y : 0.0f);
      }
    }
}

// ---- FC2 fused with q: q += relu(x1 @ w2^T + b2) . w3 ----------------------
__global__ __launch_bounds__(256)
void fc2q_kernel(const _Float16* __restrict__ ws,
                 const float* __restrict__ b2, const float* __restrict__ w3,
                 float* __restrict__ q) {
  __shared__ _Float16 sA[2][128 * KT];
  __shared__ _Float16 sB[2][128 * KT];
  const int tid = threadIdx.x, lane = tid & 31, wave = tid >> 5;
  const int mblk = blockIdx.x * 128, nblk = blockIdx.y * 128;
  const _Float16* A = ws + OFF_X1;
  const _Float16* W = ws + OFF_W2;

  const int arow = tid >> 3;
  const int aseg = tid & 7;
  unsigned ldsA0[4], ldsB0[4];
#pragma unroll
  for (int i = 0; i < 4; ++i) {
    ldsA0[i] = (unsigned)(unsigned long long)(&sA[0][swz_off(arow + 32 * i, aseg)]);
    ldsB0[i] = (unsigned)(unsigned long long)(&sB[0][swz_off(arow + 32 * i, aseg)]);
  }
  unsigned long long gA[4], gB[4];
#pragma unroll
  for (int i = 0; i < 4; ++i) {
    gA[i] = (unsigned long long)(A + (size_t)(mblk + arow + 32 * i) * HID + aseg * 8);
    gB[i] = (unsigned long long)(W + (size_t)(nblk + arow + 32 * i) * HID + aseg * 8);
  }
  auto stage = [&](int buf) {
    const unsigned ob = (unsigned)buf * (unsigned)sizeof(sA[0]);
#pragma unroll
    for (int i = 0; i < 4; ++i) { async_cp16(ldsA0[i] + ob, gA[i]); gA[i] += 2 * KT; }
#pragma unroll
    for (int i = 0; i < 4; ++i) { async_cp16(ldsB0[i] + ob, gB[i]); gB[i] += 2 * KT; }
  };

  v8f acc[2][4] = {};
  stage(0);
  async_wait0();
  __syncthreads();
  const int art = (wave >> 1) * 2;
  const int bct = (wave & 1) * 4;
  for (int kt = 0; kt < HID / KT; ++kt) {
    const int cur = kt & 1;
    if (kt + 1 < HID / KT) stage(cur ^ 1);
#pragma unroll
    for (int ks = 0; ks < 2; ++ks) {
      v16h a0 = lds_frag_a(sA[cur], art + 0, ks, lane);
      v16h a1 = lds_frag_a(sA[cur], art + 1, ks, lane);
      v16h b0 = lds_frag_b(sB[cur], bct + 0, ks, lane);
      v16h b1 = lds_frag_b(sB[cur], bct + 1, ks, lane);
      v16h b2 = lds_frag_b(sB[cur], bct + 2, ks, lane);
      v16h b3 = lds_frag_b(sB[cur], bct + 3, ks, lane);
      acc[0][0] = wmma(a0, b0, acc[0][0]);
      acc[1][0] = wmma(a1, b0, acc[1][0]);
      acc[0][1] = wmma(a0, b1, acc[0][1]);
      acc[1][1] = wmma(a1, b1, acc[1][1]);
      acc[0][2] = wmma(a0, b2, acc[0][2]);
      acc[1][2] = wmma(a1, b2, acc[1][2]);
      acc[0][3] = wmma(a0, b3, acc[0][3]);
      acc[1][3] = wmma(a1, b3, acc[1][3]);
    }
    async_wait0();
    __syncthreads();
  }
  const int m0 = mblk + (wave >> 1) * 32;
  const int n0 = nblk + (wave & 1) * 64;
  const int mo = (lane >> 4) * 8;
  const int nc = lane & 15;
  float bb[4], wv[4];
#pragma unroll
  for (int c = 0; c < 4; ++c) {
    const int n = n0 + c * 16 + nc;
    bb[c] = b2[n];
    wv[c] = w3[n];
  }
#pragma unroll
  for (int rt = 0; rt < 2; ++rt) {
#pragma unroll
    for (int v = 0; v < 8; ++v) {
      float p = 0.0f;
#pragma unroll
      for (int c = 0; c < 4; ++c) {
        float y = acc[rt][c][v] + bb[c];
        y = y > 0.0f ? y : 0.0f;
        p += y * wv[c];
      }
      p += __shfl_xor(p, 1);
      p += __shfl_xor(p, 2);
      p += __shfl_xor(p, 4);
      p += __shfl_xor(p, 8);
      if (nc == 0) {
        const int m = m0 + rt * 16 + mo + v;
        atomicAdd(&q[m], p);
      }
    }
  }
}

// ---------------------------------------------------------------------------
extern "C" void kernel_launch(void* const* d_in, const int* in_sizes, int n_in,
                              void* d_out, int out_size, void* d_ws, size_t ws_size,
                              hipStream_t stream) {
  const float* state  = (const float*)d_in[0];
  const float* action = (const float*)d_in[1];
  const float* hidden = (const float*)d_in[2];
  const float* w_ih   = (const float*)d_in[3];
  const float* w_hh   = (const float*)d_in[4];
  const float* b_ih   = (const float*)d_in[5];
  const float* b_hh   = (const float*)d_in[6];
  const float* w1     = (const float*)d_in[7];
  const float* b1     = (const float*)d_in[8];
  const float* w2     = (const float*)d_in[9];
  const float* b2     = (const float*)d_in[10];
  const float* w3     = (const float*)d_in[11];
  const float* b3     = (const float*)d_in[12];

  float* out   = (float*)d_out;
  float* q     = out;            // 65536
  float* h_out = out + BATCH;    // 65536*512
  _Float16* ws = (_Float16*)d_ws;

  auto nb = [](size_t n) { return (unsigned)((n + 255) / 256); };

  cvt_f32_f16<<<nb(1536ull * 320), 256, 0, stream>>>(w_ih, ws + OFF_WIH, 1536ull * 320);
  cvt_f32_f16<<<nb(1536ull * 512), 256, 0, stream>>>(w_hh, ws + OFF_WHH, 1536ull * 512);
  cvt_f32_f16<<<nb(1024ull * 512), 256, 0, stream>>>(w1, ws + OFF_W1, 1024ull * 512);
  cvt_f32_f16<<<nb(1024ull * 1024), 256, 0, stream>>>(w2, ws + OFF_W2, 1024ull * 1024);
  cvt_f32_f16<<<nb((size_t)BATCH * GRU), 256, 0, stream>>>(hidden, ws + OFF_H,
                                                           (size_t)BATCH * GRU);
  build_sa16<<<nb((size_t)BATCH * IN_DIM), 256, 0, stream>>>(state, action, ws + OFF_SA);
  init_q<<<nb(BATCH), 256, 0, stream>>>(q, b3);

  gru_kernel<<<dim3(BATCH / 128, GRU / 32), 256, 0, stream>>>(
      ws, hidden, b_ih, b_hh, h_out, ws + OFF_HN);
  fc1_kernel<<<dim3(BATCH / 128, HID / 128), 256, 0, stream>>>(ws, b1, ws + OFF_X1);
  fc2q_kernel<<<dim3(BATCH / 128, HID / 128), 256, 0, stream>>>(ws, b2, w3, q);
}
